// EnhancedPatchOptimizedModel_23441931502225
// MI455X (gfx1250) — compile-verified, matrix operand order checked by reading, not run on hardware
//
#include <hip/hip_runtime.h>

// ---------------------------------------------------------------------------
// Model constants (EnhancedPatchOptimizedModel)
// ---------------------------------------------------------------------------
constexpr int BB   = 128;   // batch
constexpr int NP   = 9;     // patches
constexpr int Dm   = 2048;  // feature dim
constexpr int Hn   = 4;     // heads
constexpr int DKh  = 512;   // head dim
constexpr int HIDm = 1024;  // classifier hidden
constexpr int NCm  = 101;   // classes
constexpr float EPSm = 1e-5f;

// ---------------------------------------------------------------------------
// bf16 WMMA types
// ---------------------------------------------------------------------------
typedef __bf16 bf16_t;
typedef __attribute__((ext_vector_type(4)))  bf16_t v4bf;
typedef __attribute__((ext_vector_type(8)))  bf16_t v8bf;
typedef __attribute__((ext_vector_type(16))) bf16_t v16bf;
typedef __attribute__((ext_vector_type(8)))  float  v8f;

union FragBF { v16bf v; v8bf h[2]; };

#define TILE_M 128
#define TILE_N 128
#define TILE_K 32

// ---------------------------------------------------------------------------
// GEMM: C[M,N] = act(A[M,K] @ W[K,N] + bias)   (f32 in/out, bf16 WMMA compute)
// 256 threads = 8 waves (4x2), wave tile 32x64 = 2x4 WMMA 16x16 accumulators.
// Requires K % 32 == 0 and M % 128 == 0 (all call sites satisfy this).
// ---------------------------------------------------------------------------
__global__ __launch_bounds__(256)
void gemm_bf16(const float* __restrict__ A, const float* __restrict__ W,
               const float* __restrict__ bias, float* __restrict__ C,
               int M, int N, int K, int relu)
{
    __shared__ bf16_t sA[2][TILE_M][TILE_K];   // [m][k]   8KB x2
    __shared__ bf16_t sB[2][TILE_N][TILE_K];   // [n][k]   8KB x2 (W transposed)

    const int tid  = threadIdx.x;
    const int lane = tid & 31;
    const int wave = tid >> 5;
    const int l16  = lane & 15;
    const int hi   = lane >> 4;       // 0: K{0..7,16..23}  1: K{8..15,24..31}
    const int wm   = wave & 3;        // wave row (32 rows each)
    const int wn   = wave >> 2;       // wave col (64 cols each)
    const int bN0  = blockIdx.x * TILE_N;
    const int bM0  = blockIdx.y * TILE_M;
    const bool n4  = (N & 3) == 0;

    float4 ra[4], rb[4];              // global staging registers

    auto loadG = [&](int kk) {
        #pragma unroll
        for (int u = 0; u < 4; ++u) {            // A: 128x32 f32 = 1024 float4
            int id = tid + 256 * u;
            int r = id >> 3, c4 = id & 7;
            ra[u] = *(const float4*)(A + (size_t)(bM0 + r) * K + kk + c4 * 4);
        }
        #pragma unroll
        for (int u = 0; u < 4; ++u) {            // W: 32x128 f32 = 1024 float4
            int id = tid + 256 * u;
            int kr = id >> 5, c4 = id & 31;
            int col = bN0 + c4 * 4;
            const float* wp = W + (size_t)(kk + kr) * N + col;
            float4 wv;
            if (n4 && (col + 3 < N)) {
                wv = *(const float4*)wp;
            } else {
                wv.x = (col + 0 < N) ? wp[0] : 0.f;
                wv.y = (col + 1 < N) ? wp[1] : 0.f;
                wv.z = (col + 2 < N) ? wp[2] : 0.f;
                wv.w = (col + 3 < N) ? wp[3] : 0.f;
            }
            rb[u] = wv;
        }
    };

    auto storeL = [&](int buf) {
        #pragma unroll
        for (int u = 0; u < 4; ++u) {
            int id = tid + 256 * u;
            int r = id >> 3, c4 = id & 7;
            v4bf p;
            p.x = (bf16_t)ra[u].x; p.y = (bf16_t)ra[u].y;
            p.z = (bf16_t)ra[u].z; p.w = (bf16_t)ra[u].w;
            *(v4bf*)&sA[buf][r][c4 * 4] = p;                 // 8B LDS store
        }
        #pragma unroll
        for (int u = 0; u < 4; ++u) {                        // transpose W tile
            int id = tid + 256 * u;
            int kr = id >> 5, c4 = id & 31;
            sB[buf][c4 * 4 + 0][kr] = (bf16_t)rb[u].x;
            sB[buf][c4 * 4 + 1][kr] = (bf16_t)rb[u].y;
            sB[buf][c4 * 4 + 2][kr] = (bf16_t)rb[u].z;
            sB[buf][c4 * 4 + 3][kr] = (bf16_t)rb[u].w;
        }
    };

    v8f acc[2][4];
    #pragma unroll
    for (int mi = 0; mi < 2; ++mi)
        #pragma unroll
        for (int ni = 0; ni < 4; ++ni)
            #pragma unroll
            for (int e = 0; e < 8; ++e) acc[mi][ni][e] = 0.f;

    const int ksteps = K / TILE_K;
    loadG(0);
    storeL(0);
    __syncthreads();

    for (int t = 0; t < ksteps; ++t) {
        const int buf = t & 1;
        if (t + 1 < ksteps) {
            loadG((t + 1) * TILE_K);
            if (t + 2 < ksteps) {   // global_prefetch_b8 two tiles ahead
                __builtin_prefetch(A + (size_t)(bM0 + (tid >> 3)) * K + (t + 2) * TILE_K, 0, 3);
                __builtin_prefetch(W + (size_t)((t + 2) * TILE_K + (tid >> 5)) * N + bN0, 0, 3);
            }
        }

        FragBF fa[2];
        #pragma unroll
        for (int mi = 0; mi < 2; ++mi) {
            int row = wm * 32 + mi * 16 + l16;
            fa[mi].h[0] = *(const v8bf*)&sA[buf][row][hi * 8];        // K 0..7 / 8..15
            fa[mi].h[1] = *(const v8bf*)&sA[buf][row][16 + hi * 8];   // K 16..23 / 24..31
        }
        #pragma unroll
        for (int ni = 0; ni < 4; ++ni) {
            int coln = wn * 64 + ni * 16 + l16;
            FragBF fb;
            fb.h[0] = *(const v8bf*)&sB[buf][coln][hi * 8];
            fb.h[1] = *(const v8bf*)&sB[buf][coln][16 + hi * 8];
            #pragma unroll
            for (int mi = 0; mi < 2; ++mi) {
                acc[mi][ni] = __builtin_amdgcn_wmma_f32_16x16x32_bf16(
                    false, fa[mi].v, false, fb.v, (short)0, acc[mi][ni], false, false);
            }
        }

        if (t + 1 < ksteps) storeL(buf ^ 1);
        __syncthreads();
    }

    // Epilogue: D layout — VGPR r: lanes 0-15 -> M=r, lanes 16-31 -> M=r+8; N=lane&15
    #pragma unroll
    for (int ni = 0; ni < 4; ++ni) {
        int col = bN0 + wn * 64 + ni * 16 + l16;
        if (col >= N) continue;
        float bv = bias ? bias[col] : 0.f;
        #pragma unroll
        for (int mi = 0; mi < 2; ++mi) {
            int rowBase = bM0 + wm * 32 + mi * 16 + hi * 8;
            #pragma unroll
            for (int r = 0; r < 8; ++r) {
                int row = rowBase + r;
                if (row < M) {
                    float v = acc[mi][ni][r] + bv;
                    if (relu) v = fmaxf(v, 0.f);
                    C[(size_t)row * N + col] = v;
                }
            }
        }
    }
}

// ---------------------------------------------------------------------------
// Block reduction (blockDim.x == 256)
// ---------------------------------------------------------------------------
__device__ inline float blockReduceSum256(float v, float* sh)
{
    int tid = threadIdx.x;
    sh[tid] = v; __syncthreads();
    for (int off = 128; off > 0; off >>= 1) {
        if (tid < off) sh[tid] += sh[tid + off];
        __syncthreads();
    }
    float r = sh[0]; __syncthreads();
    return r;
}

// ---------------------------------------------------------------------------
// Pairwise relation score: one block per (b,i,j); diagonal forced to exact 0
// ---------------------------------------------------------------------------
__global__ __launch_bounds__(256)
void relation_scores(const float* __restrict__ a, const float* __restrict__ b,
                     const float* __restrict__ b_rel1, const float* __restrict__ w_rel2,
                     const float* __restrict__ b_rel2, float* __restrict__ scores)
{
    __shared__ float sh[256];
    int idx = blockIdx.x;
    int j = idx % NP, i = (idx / NP) % NP, bb = idx / (NP * NP);
    if (i == j) { if (threadIdx.x == 0) scores[idx] = 0.f; return; }
    const float* ai = a + ((size_t)bb * NP + i) * Dm;
    const float* bj = b + ((size_t)bb * NP + j) * Dm;
    float s = 0.f;
    for (int d = threadIdx.x; d < Dm; d += 256) {
        float h = fmaxf(ai[d] + bj[d] + b_rel1[d], 0.f);
        s += h * w_rel2[d];
    }
    s = blockReduceSum256(s, sh);
    if (threadIdx.x == 0) scores[idx] = s + b_rel2[0];
}

// ---------------------------------------------------------------------------
// Softmax over j (9 entries) per (b,i)
// ---------------------------------------------------------------------------
__global__ void softmax9(const float* __restrict__ s, float* __restrict__ w)
{
    int g = blockIdx.x * blockDim.x + threadIdx.x;
    if (g >= BB * NP) return;
    const float* r = s + (size_t)g * NP;
    float m = r[0];
    for (int j = 1; j < NP; ++j) m = fmaxf(m, r[j]);
    float e[NP]; float sum = 0.f;
    for (int j = 0; j < NP; ++j) { e[j] = __expf(r[j] - m); sum += e[j]; }
    float inv = 1.f / sum;
    float* o = w + (size_t)g * NP;
    for (int j = 0; j < NP; ++j) o[j] = e[j] * inv;
}

// ---------------------------------------------------------------------------
// ctx[b,i,:] = sum_j relw[b,i,j] * x[b,j,:]
// ---------------------------------------------------------------------------
__global__ __launch_bounds__(256)
void ctx_wsum(const float* __restrict__ rw, const float* __restrict__ x,
              float* __restrict__ ctx)
{
    int bi = blockIdx.x;
    int b  = bi / NP;
    float w[NP];
    for (int j = 0; j < NP; ++j) w[j] = rw[(size_t)bi * NP + j];
    const float* xb = x + (size_t)b * NP * Dm;
    float* o = ctx + (size_t)bi * Dm;
    for (int d = threadIdx.x; d < Dm; d += 256) {
        float s = 0.f;
        for (int j = 0; j < NP; ++j) s += w[j] * xb[(size_t)j * Dm + d];
        o[d] = s;
    }
}

// ---------------------------------------------------------------------------
// xe = x + LayerNorm(ctxp)*g + b  (one block per row)
// ---------------------------------------------------------------------------
__global__ __launch_bounds__(256)
void ln_residual(const float* __restrict__ x, const float* __restrict__ c,
                 const float* __restrict__ g, const float* __restrict__ bta,
                 float* __restrict__ xe)
{
    __shared__ float sh[256];
    int bi = blockIdx.x;
    const float* cr = c + (size_t)bi * Dm;
    const float* xr = x + (size_t)bi * Dm;
    float s = 0.f, s2 = 0.f;
    for (int d = threadIdx.x; d < Dm; d += 256) { float v = cr[d]; s += v; s2 += v * v; }
    s  = blockReduceSum256(s,  sh);
    s2 = blockReduceSum256(s2, sh);
    float mean = s / Dm;
    float var  = s2 / Dm - mean * mean;
    float rs   = rsqrtf(var + EPSm);
    float* o = xe + (size_t)bi * Dm;
    for (int d = threadIdx.x; d < Dm; d += 256)
        o[d] = xr[d] + (cr[d] - mean) * rs * g[d] + bta[d];
}

// ---------------------------------------------------------------------------
// Attention: one block per (b,h,i). N=9, DK=512. Writes att row + output.
// ---------------------------------------------------------------------------
__global__ __launch_bounds__(256)
void attention(const float* __restrict__ q, const float* __restrict__ k,
               const float* __restrict__ v, float* __restrict__ att,
               float* __restrict__ attv)
{
    __shared__ float sh[256];
    __shared__ float aw[NP];
    int idx = blockIdx.x;
    int i = idx % NP, h = (idx / NP) % Hn, b = idx / (NP * Hn);
    const float* qr = q + ((size_t)b * NP + i) * Dm + h * DKh;

    float part[NP];
    for (int j = 0; j < NP; ++j) part[j] = 0.f;
    for (int d = threadIdx.x; d < DKh; d += 256) {
        float qd = qr[d];
        for (int j = 0; j < NP; ++j)
            part[j] += qd * k[((size_t)b * NP + j) * Dm + h * DKh + d];
    }
    const float scale = 0.044194173824159216f;   // 1/sqrt(512)
    for (int j = 0; j < NP; ++j) {
        float t = blockReduceSum256(part[j], sh);
        if (threadIdx.x == 0) aw[j] = t * scale;
    }
    __syncthreads();
    if (threadIdx.x == 0) {
        float m = aw[0];
        for (int j = 1; j < NP; ++j) m = fmaxf(m, aw[j]);
        float sum = 0.f;
        for (int j = 0; j < NP; ++j) { aw[j] = __expf(aw[j] - m); sum += aw[j]; }
        float inv = 1.f / sum;
        for (int j = 0; j < NP; ++j) { aw[j] *= inv; att[(size_t)idx * NP + j] = aw[j]; }
    }
    __syncthreads();
    float* o = attv + ((size_t)b * NP + i) * Dm + h * DKh;
    for (int d = threadIdx.x; d < DKh; d += 256) {
        float s = 0.f;
        for (int j = 0; j < NP; ++j) s += aw[j] * v[((size_t)b * NP + j) * Dm + h * DKh + d];
        o[d] = s;
    }
}

// ---------------------------------------------------------------------------
// importance = mean over (h,i) of att; gw = softmax_j(importance)
// ---------------------------------------------------------------------------
__global__ void importance_gw(const float* __restrict__ att, float* __restrict__ gw)
{
    int b = blockIdx.x * blockDim.x + threadIdx.x;
    if (b >= BB) return;
    float imp[NP];
    for (int j = 0; j < NP; ++j) imp[j] = 0.f;
    for (int h = 0; h < Hn; ++h)
        for (int i = 0; i < NP; ++i) {
            const float* r = att + (((size_t)b * Hn + h) * NP + i) * NP;
            for (int j = 0; j < NP; ++j) imp[j] += r[j];
        }
    const float inv36 = 1.f / (Hn * NP);
    float m = -1e30f;
    for (int j = 0; j < NP; ++j) { imp[j] *= inv36; m = fmaxf(m, imp[j]); }
    float sum = 0.f;
    for (int j = 0; j < NP; ++j) { imp[j] = __expf(imp[j] - m); sum += imp[j]; }
    float inv = 1.f / sum;
    for (int j = 0; j < NP; ++j) gw[(size_t)b * NP + j] = imp[j] * inv;
}

// ---------------------------------------------------------------------------
// agg[b,:] = sum_n outp[b,n,:] * gw[b,n]
// ---------------------------------------------------------------------------
__global__ __launch_bounds__(256)
void pool_agg(const float* __restrict__ outp, const float* __restrict__ gw,
              float* __restrict__ agg)
{
    int b = blockIdx.x;
    float w[NP];
    for (int j = 0; j < NP; ++j) w[j] = gw[(size_t)b * NP + j];
    const float* ob = outp + (size_t)b * NP * Dm;
    float* o = agg + (size_t)b * Dm;
    for (int d = threadIdx.x; d < Dm; d += 256) {
        float s = 0.f;
        for (int j = 0; j < NP; ++j) s += w[j] * ob[(size_t)j * Dm + d];
        o[d] = s;
    }
}

// ---------------------------------------------------------------------------
// BatchNorm1d (eval) + ReLU
// ---------------------------------------------------------------------------
__global__ void bn_relu(const float* __restrict__ y, const float* __restrict__ g,
                        const float* __restrict__ b2, const float* __restrict__ m,
                        const float* __restrict__ vv, float* __restrict__ o)
{
    int idx = blockIdx.x * blockDim.x + threadIdx.x;
    if (idx >= BB * HIDm) return;
    int hc = idx % HIDm;
    float val = (y[idx] - m[hc]) * rsqrtf(vv[hc] + EPSm) * g[hc] + b2[hc];
    o[idx] = fmaxf(val, 0.f);
}

// ---------------------------------------------------------------------------
// Launch
// ---------------------------------------------------------------------------
extern "C" void kernel_launch(void* const* d_in, const int* in_sizes, int n_in,
                              void* d_out, int out_size, void* d_ws, size_t ws_size,
                              hipStream_t stream)
{
    (void)in_sizes; (void)n_in; (void)out_size; (void)ws_size;

    const float* pf     = (const float*)d_in[0];
    const float* wda1   = (const float*)d_in[1];
    const float* bda1   = (const float*)d_in[2];
    const float* wda2   = (const float*)d_in[3];
    const float* bda2   = (const float*)d_in[4];
    const float* w_rel1 = (const float*)d_in[5];
    const float* b_rel1 = (const float*)d_in[6];
    const float* w_rel2 = (const float*)d_in[7];
    const float* b_rel2 = (const float*)d_in[8];
    const float* w_ctx  = (const float*)d_in[9];
    const float* b_ctx  = (const float*)d_in[10];
    const float* ln_g   = (const float*)d_in[11];
    const float* ln_b   = (const float*)d_in[12];
    const float* wq     = (const float*)d_in[13];
    const float* bq     = (const float*)d_in[14];
    const float* wk     = (const float*)d_in[15];
    const float* bk     = (const float*)d_in[16];
    const float* wv     = (const float*)d_in[17];
    const float* bv     = (const float*)d_in[18];
    const float* wo     = (const float*)d_in[19];
    const float* bo     = (const float*)d_in[20];
    const float* wc1    = (const float*)d_in[21];
    const float* bc1    = (const float*)d_in[22];
    const float* bn_g   = (const float*)d_in[23];
    const float* bn_b   = (const float*)d_in[24];
    const float* bn_m   = (const float*)d_in[25];
    const float* bn_v   = (const float*)d_in[26];
    const float* wc2    = (const float*)d_in[27];
    const float* bc2    = (const float*)d_in[28];

    char* ws = (char*)d_ws;
    const size_t BIG = (size_t)BB * NP * Dm * sizeof(float);   // 9.44 MB
    float* bufA = (float*)(ws);
    float* bufB = (float*)(ws + BIG);
    float* bufC = (float*)(ws + 2 * BIG);
    float* bufD = (float*)(ws + 3 * BIG);
    char* p = ws + 4 * BIG;
    float* att    = (float*)p; p += (size_t)BB * Hn * NP * NP * sizeof(float);
    float* scores = (float*)p; p += (size_t)BB * NP * NP * sizeof(float);
    float* relw   = (float*)p; p += (size_t)BB * NP * NP * sizeof(float);
    float* gw     = (float*)p; p += (size_t)BB * NP * sizeof(float);
    float* agg    = (float*)p; p += (size_t)BB * Dm * sizeof(float);
    float* y1     = (float*)p; p += (size_t)BB * HIDm * sizeof(float);
    float* y2     = (float*)p; p += (size_t)BB * HIDm * sizeof(float);

    const dim3 blk(256);
    auto gemm = [&](const float* A, const float* W, const float* bias, float* C,
                    int M, int N, int K, int relu) {
        dim3 grid((N + TILE_N - 1) / TILE_N, (M + TILE_M - 1) / TILE_M);
        gemm_bf16<<<grid, blk, 0, stream>>>(A, W, bias, C, M, N, K, relu);
    };

    const int M = BB * NP;   // 1152

    // domain adaptation MLP
    gemm(pf,   wda1, bda1, bufA, M, Dm, Dm, 1);                    // t1 = relu(X@wda1+b)
    gemm(bufA, wda2, bda2, bufB, M, Dm, Dm, 0);                    // x  (bufB)

    // cross-patch relation
    gemm(bufB, w_rel1,                    nullptr, bufA, M, Dm, Dm, 0);  // a
    gemm(bufB, w_rel1 + (size_t)Dm * Dm,  nullptr, bufC, M, Dm, Dm, 0);  // b
    relation_scores<<<BB * NP * NP, blk, 0, stream>>>(bufA, bufC, b_rel1, w_rel2, b_rel2, scores);
    softmax9<<<(BB * NP + 255) / 256, blk, 0, stream>>>(scores, relw);
    ctx_wsum<<<BB * NP, blk, 0, stream>>>(relw, bufB, bufA);             // ctx_in
    gemm(bufA, w_ctx, b_ctx, bufC, M, Dm, Dm, 0);                        // ctx proj
    ln_residual<<<BB * NP, blk, 0, stream>>>(bufB, bufC, ln_g, ln_b, bufA); // xe

    // multi-head attention
    gemm(bufA, wq, bq, bufB, M, Dm, Dm, 0);                              // q
    gemm(bufA, wk, bk, bufC, M, Dm, Dm, 0);                              // k
    gemm(bufA, wv, bv, bufD, M, Dm, Dm, 0);                              // v
    attention<<<BB * Hn * NP, blk, 0, stream>>>(bufB, bufC, bufD, att, bufA);  // attv
    gemm(bufA, wo, bo, bufB, M, Dm, Dm, 0);                              // out proj
    importance_gw<<<1, 128, 0, stream>>>(att, gw);
    pool_agg<<<BB, blk, 0, stream>>>(bufB, gw, agg);

    // classifier
    gemm(agg, wc1, bc1, y1, BB, HIDm, Dm, 0);
    bn_relu<<<(BB * HIDm + 255) / 256, blk, 0, stream>>>(y1, bn_g, bn_b, bn_m, bn_v, y2);
    gemm(y2, wc2, bc2, (float*)d_out, BB, NCm, HIDm, 0);
}
